// NamedEntityRecog_79121887527581
// MI455X (gfx1250) — matrix operand order repaired and checked by reading, not compiled, not run on hardware
//
#include <hip/hip_runtime.h>
#include <hip/hip_bf16.h>

// Problem constants (from reference)
#define V_  50000
#define E_  100
#define EP_ 128      // E padded to K-multiple of 32
#define H_  256
#define G_  1024     // 4*H
#define T_  128
#define B_  512
#define NT_ 20

typedef __attribute__((ext_vector_type(16))) __bf16 v16bf;
typedef __attribute__((ext_vector_type(8)))  float  v8f;
typedef __attribute__((ext_vector_type(4))) unsigned int u32x4;
typedef __attribute__((ext_vector_type(8))) int          i32x8;
typedef __attribute__((ext_vector_type(4))) int          i32x4;

// addrspace(1)-qualified builtin vector for global 128-bit loads (forces global_load, not flat)
typedef __attribute__((address_space(1))) const u32x4 gu32x4;

union FragU { v16bf v; u32x4 u[2]; unsigned short s[16]; };

__device__ __forceinline__ float bf2f(unsigned short u) {
    union { unsigned int i; float f; } c; c.i = ((unsigned int)u) << 16; return c.f;
}
__device__ __forceinline__ unsigned short f2bf(float x) {
    union { float f; unsigned int i; } c; c.f = x;
    unsigned int r = c.i + 0x7FFFu + ((c.i >> 16) & 1u);
    return (unsigned short)(r >> 16);
}
// fast sigmoid/tanh: v_exp_f32 + v_rcp_f32 (no IEEE div sequence on the recurrent critical path)
__device__ __forceinline__ float sigf(float x) {
    return __builtin_amdgcn_rcpf(1.0f + __expf(-x));
}
__device__ __forceinline__ float tanhfast(float x) {
    float e = __expf(2.0f * x);
    return (e - 1.0f) * __builtin_amdgcn_rcpf(e + 1.0f);
}

// B-fragment from laundered global base (explicit addrspace(1) -> global_load_b128, not flat_load)
__device__ __forceinline__ v16bf loadB_g(unsigned long base, unsigned eOff, int lane) {
    const gu32x4* q = (const gu32x4*)(base + (((unsigned long)eOff + (unsigned)(lane * 16)) << 1));
    FragU f; f.u[0] = q[0]; f.u[1] = q[1]; return f.v;
}

// A-fragment from a row-major [16][ld] bf16 LDS tile (tile base pre-offset), K-chunk kt.
// ISA 16-bit A 16x32 layout: lane<16 holds K {0..7,16..23}, lane>=16 holds K {8..15,24..31}.
__device__ __forceinline__ v16bf loadA_lds(const unsigned short* sh, int ld, int kt, int lane) {
    int row = lane & 15, hi = lane >> 4;
    int c0 = kt * 32 + hi * 8;
    FragU f;
    f.u[0] = *(const u32x4*)(sh + row * ld + c0);
    f.u[1] = *(const u32x4*)(sh + row * ld + c0 + 16);
    return f.v;
}

__device__ __forceinline__ v8f wmma_bf(v16bf a, v16bf b, v8f c) {
    return __builtin_amdgcn_wmma_f32_16x16x32_bf16(false, a, false, b, (short)0, c, false, false);
}

// Tensor Data Mover: 1-D copy of one 8KB x-tile (4096 bf16) global -> LDS.
// D# group0: count=1 | lds_addr | global_addr | type=2.  group1: data_size=2B,
// tensor_dim0 = tile_dim0 = 4096, dim0_stride = 4096, no multicast/pad/iterate.
// This toolchain's builtin is the 6-arg form: (g0, g1, g2, g3, g_extra, cpol).
__device__ __forceinline__ void tdm_load_x(const unsigned short* gsrc, unsigned ldsByteOff) {
    unsigned long ga = (unsigned long)gsrc;
    unsigned lo = (unsigned)__builtin_amdgcn_readfirstlane((int)(ga & 0xFFFFFFFFu));
    unsigned hw = (unsigned)__builtin_amdgcn_readfirstlane((int)(ga >> 32));
    u32x4 g0;
    g0[0] = 1u;                                   // count=1 (valid user descriptor)
    g0[1] = ldsByteOff;                           // lds_addr (bytes)
    g0[2] = lo;                                   // global_addr[31:0]
    g0[3] = (hw & 0x01FFFFFFu) | 0x80000000u;     // global_addr[56:32] | type=2 ("image")
    i32x8 g1;
    g1[0] = 0x00010000;                           // workgroup_mask=0 | data_size=1 (2 bytes)
    g1[1] = 0x10000000;                           // tensor_dim0[15:0]=4096 in bits[63:48]
    g1[2] = 0x00010000;                           // tensor_dim0[31:16]=0 | tensor_dim1[15:0]=1
    g1[3] = 0x10000000;                           // tensor_dim1[31:16]=0 | tile_dim0=4096
    g1[4] = 0;                                    // tile_dim1=0 (unused), tile_dim2=0
    g1[5] = 4096;                                 // tensor_dim0_stride[31:0]
    g1[6] = 0;
    g1[7] = 0;
    i32x4 z4; z4[0] = 0; z4[1] = 0; z4[2] = 0; z4[3] = 0;
    i32x8 z8; z8[0] = 0; z8[1] = 0; z8[2] = 0; z8[3] = 0;
    z8[4] = 0; z8[5] = 0; z8[6] = 0; z8[7] = 0;
    __builtin_amdgcn_tensor_load_to_lds(g0, g1, z4, z4, z8, 0);
}

// ---------------- packing kernels ----------------

// xpack[t][btile(16)][r(32)][c(128)] bf16 = embed[word_inputs[btile*32+r, t], c], zero pad c>=100
__global__ __launch_bounds__(256) void embed_pack(const int* __restrict__ words,
                                                  const float* __restrict__ embed,
                                                  unsigned short* __restrict__ xpack) {
    long e = (long)blockIdx.x * 256 + threadIdx.x;        // total = T*16*32*128 = 8388608
    int cc = (int)(e & 127);
    int r  = (int)((e >> 7) & 31);
    int bt = (int)((e >> 12) & 15);
    int t  = (int)(e >> 16);
    int b  = bt * 32 + r;
    float v = 0.f;
    if (cc < E_) {
        int w = words[b * T_ + t];
        if (w >= 0 && w < V_) v = embed[(size_t)w * E_ + cc];
    }
    xpack[e] = f2bf(v);
}

// Pack a logical KxN matrix (optionally transposed source, zero-padded) into WMMA B-fragment tiles.
__global__ __launch_bounds__(256) void pack_b(const float* __restrict__ src,
                                              unsigned short* __restrict__ dst,
                                              int Ntiles, int srcInner,
                                              int kLimit, int nLimit, int transpose, int total) {
    int e = blockIdx.x * 256 + threadIdx.x;
    if (e >= total) return;
    int j    = e & 15;
    int lane = (e >> 4) & 31;
    int tile = e >> 9;
    int ntile = tile % Ntiles, kt = tile / Ntiles;
    int n = ntile * 16 + (lane & 15);
    int k = kt * 32 + (lane >> 4) * 8 + (j & 7) + (j >> 3) * 16;
    float v = 0.f;
    if (k < kLimit && n < nLimit)
        v = transpose ? src[(size_t)n * srcInner + k] : src[(size_t)k * srcInner + n];
    dst[e] = f2bf(v);
}

__global__ __launch_bounds__(256) void bias_sum(const float* __restrict__ bihf, const float* __restrict__ bhhf,
                                                const float* __restrict__ bihb, const float* __restrict__ bhhb,
                                                float* __restrict__ bsum) {
    int i = blockIdx.x * 256 + threadIdx.x;
    if (i < G_)            bsum[i] = bihf[i] + bhhf[i];
    else if (i < 2 * G_)   bsum[i] = bihb[i - G_] + bhhb[i - G_];
}

// ---------------- persistent recurrent kernel ----------------
// grid = 32 blocks: blockIdx = dir*16 + btile (M=32 batch rows per block).
// block = 512 threads = 16 waves. Wave w owns H-columns [16w,16w+16) of h/c for both M-tiles.
// Each B fragment load is reused for 2 WMMAs (two M-tiles) -> halves L2 weight traffic per FLOP.
// x tiles are DMA'd by the Tensor Data Mover into a double-buffered LDS stage (wave 0 issues,
// s_wait_tensorcnt + workgroup barrier publishes).
#define SM_H   0
#define SM_HM  8192
#define SM_XM  16384
#define SM_XS  20480              // element offsets into smem[]
#define SM_XS_BYTES 40960u        // byte offset of x stage (assumes smem at LDS offset 0)

__global__ __launch_bounds__(512) void mog_lstm_main(const unsigned short* __restrict__ xpack,
                                                     const unsigned short* __restrict__ wpack,
                                                     const float* __restrict__ bsum,
                                                     unsigned short* __restrict__ feat) {
    // single shared block so TDM lds_addr byte offsets are known constants
    __shared__ __align__(16) unsigned short smem[8192 + 8192 + 4096 + 2 * 4096]; // 56 KiB
    unsigned short* h_bf  = smem + SM_H;    // 32 x 256
    unsigned short* hm_bf = smem + SM_HM;   // 32 x 256
    unsigned short* xm_bf = smem + SM_XM;   // 32 x 128
    unsigned short* xstg  = smem + SM_XS;   // 2 x (32 x 128)

    const int dir   = blockIdx.x >> 4;
    const int btile = blockIdx.x & 15;
    const int wave  = threadIdx.x >> 5;   // 0..15
    const int lane  = threadIdx.x & 31;
    const int col   = lane & 15;
    const int hi    = lane >> 4;

    const unsigned short* Qp   = wpack + (size_t)dir * 458752u;
    const unsigned short* Rp   = Qp + 32768;
    const unsigned short* WihP = Qp + 65536;
    const unsigned short* WhhP = Qp + 196608;
    const float* bias = bsum + dir * G_;

    for (int i = threadIdx.x; i < 32 * H_; i += 512) h_bf[i] = 0;
    v8f h[2], c[2];
    #pragma unroll
    for (int m = 0; m < 2; ++m)
        for (int r = 0; r < 8; ++r) { h[m][r] = 0.f; c[m][r] = 0.f; }

    // prologue: DMA tile for t=0 into stage buffer 0
    if (wave == 0) {
        const int t0 = dir ? (T_ - 1) : 0;
        tdm_load_x(xpack + (((size_t)t0 * 16 + btile) << 12), SM_XS_BYTES);
    }

    const int mu = wave >> 3;   // step-1 job: which M-tile
    const int nu = wave & 7;    // step-1 job: which E-tile

    for (int t = 0; t < T_; ++t) {
        const int tx = dir ? (T_ - 1 - t) : t;

        // tile t is in flight (issued last iteration / prologue): wait + publish
        if (wave == 0) __builtin_amdgcn_s_wait_tensorcnt(0);
        __syncthreads();   // also orders step-4(t-1) h_bf writes vs step-1 reads

        // kick off DMA of tile t+1 into the other stage buffer
        if (wave == 0 && t + 1 < T_) {
            const int tn = dir ? (T_ - 2 - t) : (t + 1);
            tdm_load_x(xpack + (((size_t)tn * 16 + btile) << 12),
                       SM_XS_BYTES + (unsigned)(((t + 1) & 1) << 13));
        }
        const unsigned short* xcur = xstg + ((t & 1) << 12);

        // Opaque per-iteration weight bases: stops the compiler from hoisting the
        // loop-invariant B-fragment loads out of the t-loop (which spilled to scratch).
        unsigned long qa = (unsigned long)Qp,  ra = (unsigned long)Rp;
        unsigned long ia = (unsigned long)WihP, ha = (unsigned long)WhhP;
        asm volatile("" : "+s"(qa), "+s"(ra), "+s"(ia), "+s"(ha));

        // ---- step 1: u = h @ Q ; wave -> (M-tile mu, E-tile nu); xm = 2*sig(u)*x ----
        v8f u; for (int r = 0; r < 8; ++r) u[r] = 0.f;
        #pragma unroll
        for (int kt = 0; kt < 8; ++kt) {
            v16bf a = loadA_lds(h_bf + mu * 16 * H_, H_, kt, lane);
            v16bf b = loadB_g(qa, (unsigned)((kt * 8 + nu) << 9), lane);
            u = wmma_bf(a, b, u);
        }
        #pragma unroll
        for (int r = 0; r < 8; ++r) {
            int grow = mu * 16 + r + 8 * hi, n = nu * 16 + col;
            float xv = bf2f(xcur[grow * EP_ + n]);
            xm_bf[grow * EP_ + n] = f2bf(2.f * sigf(u[r]) * xv);
        }
        __syncthreads();

        // ---- step 2: v = xm @ R ; wave -> H-tile `wave`, both M-tiles; hm = 2*sig(v)*h ----
        v8f v0, v1; for (int r = 0; r < 8; ++r) { v0[r] = 0.f; v1[r] = 0.f; }
        #pragma unroll
        for (int kt = 0; kt < 4; ++kt) {
            v16bf a0 = loadA_lds(xm_bf,            EP_, kt, lane);
            v16bf a1 = loadA_lds(xm_bf + 16 * EP_, EP_, kt, lane);
            v16bf b  = loadB_g(ra, (unsigned)((kt * 16 + wave) << 9), lane);
            v0 = wmma_bf(a0, b, v0);
            v1 = wmma_bf(a1, b, v1);
        }
        #pragma unroll
        for (int r = 0; r < 8; ++r) {
            int mrow = r + 8 * hi, n = wave * 16 + col;
            hm_bf[mrow * H_ + n]        = f2bf(2.f * sigf(v0[r]) * h[0][r]);
            hm_bf[(16 + mrow) * H_ + n] = f2bf(2.f * sigf(v1[r]) * h[1][r]);
        }
        __syncthreads();

        // ---- step 3: g = xm@Wih^T + hm@Whh^T + b ; wave -> N-tiles {gb*16+wave}, 2 M-tiles ----
        v8f g[4][2];
        #pragma unroll
        for (int gb = 0; gb < 4; ++gb) {
            float bv = bias[(gb * 16 + wave) * 16 + col];
            for (int r = 0; r < 8; ++r) { g[gb][0][r] = bv; g[gb][1][r] = bv; }
        }
        #pragma unroll
        for (int kt = 0; kt < 4; ++kt) {
            v16bf a0 = loadA_lds(xm_bf,            EP_, kt, lane);
            v16bf a1 = loadA_lds(xm_bf + 16 * EP_, EP_, kt, lane);
            #pragma unroll
            for (int gb = 0; gb < 4; ++gb) {
                v16bf b = loadB_g(ia, (unsigned)((kt * 64 + gb * 16 + wave) << 9), lane);
                g[gb][0] = wmma_bf(a0, b, g[gb][0]);
                g[gb][1] = wmma_bf(a1, b, g[gb][1]);
            }
        }
        #pragma unroll
        for (int kt = 0; kt < 8; ++kt) {
            v16bf a0 = loadA_lds(hm_bf,           H_, kt, lane);
            v16bf a1 = loadA_lds(hm_bf + 16 * H_, H_, kt, lane);
            #pragma unroll
            for (int gb = 0; gb < 4; ++gb) {
                v16bf b = loadB_g(ha, (unsigned)((kt * 64 + gb * 16 + wave) << 9), lane);
                g[gb][0] = wmma_bf(a0, b, g[gb][0]);
                g[gb][1] = wmma_bf(a1, b, g[gb][1]);
            }
        }

        // ---- step 4: LSTM gates (i,f,g,o), update h/c, publish h ----
        unsigned short* fout = feat + ((((size_t)tx * 2 + dir) * 16 + btile) << 13); // 32x256 tile
        #pragma unroll
        for (int m = 0; m < 2; ++m) {
            v8f gi = g[0][m], gf = g[1][m], gg = g[2][m], go = g[3][m];
            #pragma unroll
            for (int r = 0; r < 8; ++r) {
                float cv = sigf(gf[r]) * c[m][r] + sigf(gi[r]) * tanhfast(gg[r]);
                c[m][r] = cv;
                float hv = sigf(go[r]) * tanhfast(cv);
                h[m][r] = hv;
                int grow = m * 16 + r + 8 * hi, n = wave * 16 + col;
                unsigned short hb16 = f2bf(hv);
                h_bf[grow * H_ + n] = hb16;
                fout[grow * H_ + n] = hb16;
            }
        }
        // no trailing barrier: top-of-loop barrier orders h_bf writes vs next step-1 reads
    }
}

// ---------------- logits + argmax + mask ----------------
__global__ __launch_bounds__(256) void logits_argmax(const unsigned short* __restrict__ feat,
                                                     const float* __restrict__ Wt,
                                                     const float* __restrict__ bt,
                                                     const unsigned char* __restrict__ mask,
                                                     int* __restrict__ out) {
    int idx = blockIdx.x * 256 + threadIdx.x;
    if (idx >= T_ * B_) return;
    int t = idx / B_, b = idx % B_;
    int btile = b >> 5, r = b & 31;
    const unsigned short* hf = feat + ((((size_t)t * 2 + 0) * 16 + btile) << 13) + r * H_;
    const unsigned short* hb = feat + ((((size_t)t * 2 + 1) * 16 + btile) << 13) + r * H_;
    float best = -3.4e38f; int arg = 0;
    for (int cl = 0; cl < NT_; ++cl) {
        const float* w = Wt + cl * (2 * H_);
        float acc = bt[cl];
        for (int k = 0; k < H_; ++k) acc += bf2f(hf[k]) * w[k];
        for (int k = 0; k < H_; ++k) acc += bf2f(hb[k]) * w[H_ + k];
        if (acc > best) { best = acc; arg = cl; }
    }
    out[b * T_ + t] = mask[b * T_ + t] ? arg : 0;
}

extern "C" void kernel_launch(void* const* d_in, const int* in_sizes, int n_in,
                              void* d_out, int out_size, void* d_ws, size_t ws_size,
                              hipStream_t stream) {
    (void)in_sizes; (void)n_in; (void)out_size; (void)ws_size;

    const int*           words = (const int*)d_in[0];
    const unsigned char* mask  = (const unsigned char*)d_in[4];
    const float* embed = (const float*)d_in[5];
    const float* Wih_f = (const float*)d_in[6];
    const float* Whh_f = (const float*)d_in[7];
    const float* bih_f = (const float*)d_in[8];
    const float* bhh_f = (const float*)d_in[9];
    const float* Q_f   = (const float*)d_in[10];
    const float* R_f   = (const float*)d_in[11];
    const float* Wih_b = (const float*)d_in[12];
    const float* Whh_b = (const float*)d_in[13];
    const float* bih_b = (const float*)d_in[14];
    const float* bhh_b = (const float*)d_in[15];
    const float* Q_b   = (const float*)d_in[16];
    const float* R_b   = (const float*)d_in[17];
    const float* Wt    = (const float*)d_in[18];
    const float* bt    = (const float*)d_in[19];

    // workspace layout
    unsigned short* xpack = (unsigned short*)d_ws;                                      // 16 MiB
    unsigned short* wpack = (unsigned short*)((char*)d_ws + 16777216);                  // 1.75 MiB
    float*          bsum  = (float*)((char*)d_ws + 16777216 + 1835008);                 // 8 KiB
    unsigned short* feat  = (unsigned short*)((char*)d_ws + 16777216 + 1835008 + 8192); // 64 MiB

    // 1) embedding gather -> padded bf16 tiles [t][btile16][32][128]
    embed_pack<<<(T_ * 16 * 32 * 128) / 256, 256, 0, stream>>>(words, embed, xpack);

    // 2) pack weights into WMMA B-fragment order, bf16 (per direction)
    const float* Qs[2]   = { Q_f,   Q_b   };
    const float* Rs[2]   = { R_f,   R_b   };
    const float* Wihs[2] = { Wih_f, Wih_b };
    const float* Whhs[2] = { Whh_f, Whh_b };
    for (int d = 0; d < 2; ++d) {
        unsigned short* base = wpack + (size_t)d * 458752u;
        // Q: K=256, N=128(pad from 100), src [H][E] direct
        pack_b<<<32768 / 256, 256, 0, stream>>>(Qs[d],    base +      0,  8, E_, 256, E_, 0, 32768);
        // R: K=128(pad from 100), N=256, src [E][H] direct
        pack_b<<<32768 / 256, 256, 0, stream>>>(Rs[d],    base +  32768, 16, H_, E_,  H_, 0, 32768);
        // Wih^T: K=128(pad), N=1024, src [1024][E] transposed
        pack_b<<<131072 / 256, 256, 0, stream>>>(Wihs[d], base +  65536, 64, E_, E_,  G_, 1, 131072);
        // Whh^T: K=256, N=1024, src [1024][H] transposed
        pack_b<<<262144 / 256, 256, 0, stream>>>(Whhs[d], base + 196608, 64, H_, H_,  G_, 1, 262144);
    }

    // 3) combined biases
    bias_sum<<<8, 256, 0, stream>>>(bih_f, bhh_f, bih_b, bhh_b, bsum);

    // 4) persistent bidirectional Mogrifier-LSTM (32 blocks: 2 dirs x 16 batch tiles of M=32)
    mog_lstm_main<<<32, 512, 0, stream>>>(xpack, wpack, bsum, feat);

    // 5) logits + argmax + mask
    logits_argmax<<<(T_ * B_) / 256, 256, 0, stream>>>(feat, Wt, bt, mask, (int*)d_out);
}
